// MPNN_46334107189441
// MI455X (gfx1250) — matrix operand order repaired and checked by reading, not compiled, not run on hardware
//
#include <hip/hip_runtime.h>
#include <hip/hip_bf16.h>

// MPNN on MI455X (gfx1250): all dense GEMMs on v_wmma_f32_16x16x32_f16 with
// fp16 operands pre-staged in WMMA-friendly layouts (A row-major, B col-major),
// f32 accumulation; nonlinear per-edge reduction stays VALU+LDS.

typedef __attribute__((ext_vector_type(16))) _Float16 v16h;
typedef __attribute__((ext_vector_type(8)))  _Float16 v8h;
typedef __attribute__((ext_vector_type(8)))  float    v8f;
typedef _Float16 f16;

#define NB 2
#define NN 1024
#define NU 512
#define NF 64
#define NL 4

// ---- WMMA f16 step: c += A(16x32) * B(32x16), A and B share one k order ----
// A per lane m+16h: element e -> K=16h+e
//   => one contiguous 16-f16 run at k0+16h in row m (A row-major).
// B per lane n+16h: element e -> K=16h+e
//   => one contiguous 16-f16 run at k0+16h in column n (B col-major).

// ---------------- prep: weights -> f16 (direct + transposed) ----------------
__global__ void k_prep_w(const float* __restrict__ wts, f16* __restrict__ w16,
                         f16* __restrict__ w16T)
{
    int tid = blockIdx.x * blockDim.x + threadIdx.x;      // over 2*512*512
    if (tid >= NB * NU * NU) return;
    w16[tid] = (f16)wts[tid];
    int b = tid >> 18, r = tid & (NU * NU - 1);
    int i = r >> 9, k = r & (NU - 1);
    w16T[(size_t)b * NU * NU + (size_t)i * NU + k] = (f16)wts[(size_t)b * NU * NU + (size_t)k * NU + i];
}

// ------- prep: Wef/Wmsg/Wupd -> f16 column-major [n][k] -------
__global__ void k_prep_small(const float* __restrict__ Wef, const float* __restrict__ Wmsg,
                             const float* __restrict__ Wupd, f16* __restrict__ WefT,
                             f16* __restrict__ WmsgT, f16* __restrict__ WupdT)
{
    int tid = blockIdx.x * blockDim.x + threadIdx.x;
    if (tid < 4096) {                                     // Wef 64x64
        int n = tid >> 6, k = tid & 63;
        WefT[n * 64 + k] = (f16)Wef[k * 64 + n];
    } else if (tid < 4096 + 32768) {                      // Wmsg (4,128,64)
        int t = tid - 4096;
        int l = t >> 13, r = t & 8191, n = r >> 7, k = r & 127;
        WmsgT[l * 8192 + n * 128 + k] = (f16)Wmsg[l * 8192 + k * 64 + n];
    } else if (tid < 4096 + 65536) {                      // Wupd (4,128,64)
        int t = tid - 4096 - 32768;
        int l = t >> 13, r = t & 8191, n = r >> 7, k = r & 127;
        WupdT[l * 8192 + n * 128 + k] = (f16)Wupd[l * 8192 + k * 64 + n];
    }
}

// ---------------- x0 = relu(nf * W_init + b_init), f16 row-major + transposed ----------------
__global__ void k_x0(const float* __restrict__ nf, const float* __restrict__ Wi,
                     const float* __restrict__ bi, f16* __restrict__ x16,
                     f16* __restrict__ xT16)
{
    int idx = blockIdx.x * blockDim.x + threadIdx.x;      // over NB*NN*NF
    if (idx >= NB * NN * NF) return;
    int f = idx & (NF - 1);
    int bn = idx >> 6;
    int b = bn >> 10, n = bn & (NN - 1);
    float v = fmaxf(nf[bn] * Wi[f] + bi[f], 0.0f);
    x16[idx] = (f16)v;
    xT16[((size_t)b * NF + f) * NN + n] = (f16)v;
}

// ---------------- norm[b,i] = max(1, sum_j [adj[b,j,i]==0]) ----------------
__global__ void k_norm(const float* __restrict__ adj, float* __restrict__ norms)
{
    int tid = blockIdx.x * blockDim.x + threadIdx.x;      // over NB*NN
    if (tid >= NB * NN) return;
    int b = tid >> 10, i = tid & (NN - 1);
    const float* ab = adj + (size_t)b * NN * NN;
    float cnt = 0.0f;
    for (int j = 0; j < NN; ++j)                          // coalesced across lanes
        cnt += (ab[(size_t)j * NN + i] == 0.0f) ? 1.0f : 0.0f;
    norms[tid] = fmaxf(cnt, 1.0f);
}

__global__ void k_normmax(const float* __restrict__ norms, float* __restrict__ nmax)
{
    __shared__ float red[256];
    float m = 0.0f;
    for (int i = threadIdx.x; i < NB * NN; i += 256) m = fmaxf(m, norms[i]);
    red[threadIdx.x] = m;
    __syncthreads();
    for (int s = 128; s > 0; s >>= 1) {
        if (threadIdx.x < s) red[threadIdx.x] = fmaxf(red[threadIdx.x], red[threadIdx.x + s]);
        __syncthreads();
    }
    if (threadIdx.x == 0) nmax[0] = red[0];
}

// ------- eein[b,i,k<63] = (sum_j mask*relu(Wfull*W0k + nf_j*W1k))/norm ; k=63 -> norm/nmax
__global__ void k_edgesum(const float* __restrict__ adj, const float* __restrict__ wts,
                          const float* __restrict__ nf,  const float* __restrict__ Wee,
                          const float* __restrict__ norms, const float* __restrict__ nmax,
                          f16* __restrict__ eein16)
{
    const int bi = blockIdx.x;                            // b*NN+i
    const int b = bi >> 10, i = bi & (NN - 1);
    const int t = threadIdx.x;                            // 64 threads
    __shared__ float s_adj[64], s_w[64], s_nf[64];
    const float* arow = adj + (size_t)bi * NN;            // mask index [b,i,j]
    const float* wb = wts + (size_t)b * NU * NU;
    const float* nfb = nf + (size_t)b * NN;
    float w0 = 0.0f, w1 = 0.0f;
    if (t < 63) { w0 = Wee[t]; w1 = Wee[63 + t]; }
    float acc = 0.0f;
    for (int j0 = 0; j0 < NN; j0 += 64) {
        int j = j0 + t;
        if (j0 + 64 < NN) __builtin_prefetch(arow + j0 + 64 + t, 0, 1);
        s_adj[t] = arow[j];
        s_nf[t]  = nfb[j];
        float wv;
        if (i < NU) wv = (j >= NU) ? wb[(size_t)(j - NU) * NU + i] : 0.0f;
        else        wv = (j <  NU) ? wb[(size_t)(i - NU) * NU + j] : 0.0f;
        s_w[t] = wv;
        __syncthreads();
        if (t < 63) {
            #pragma unroll 8
            for (int jj = 0; jj < 64; ++jj) {
                if (s_adj[jj] == 0.0f)
                    acc += fmaxf(s_w[jj] * w0 + s_nf[jj] * w1, 0.0f);
            }
        }
        __syncthreads();
    }
    float nrm = norms[bi];
    if (t < 63)  eein16[(size_t)bi * NF + t] = (f16)(acc / nrm);
    if (t == 63) eein16[(size_t)bi * NF + 63] = (f16)(nrm / nmax[0]);
}

// ------- out(Mx64) = act( [P | Q](MxK) @ W(Kx64) ); P/Q f16 row-major, WT f16 col-major
// 128 threads = 4 waves; each wave: one 16-row strip, all 64 cols (4 C tiles).
__global__ void k_gemm_concat(const f16* __restrict__ P, const f16* __restrict__ Q,
                              const f16* __restrict__ WT, float* __restrict__ out32,
                              f16* __restrict__ out16, f16* __restrict__ out16T,
                              int relu)
{
    const int lane = threadIdx.x & 31;
    const int h = lane >> 4, mr = lane & 15;
    const int m0 = (blockIdx.x * 4 + (threadIdx.x >> 5)) * 16;
    const int K = Q ? 128 : 64;
    v8f acc[4] = {{}, {}, {}, {}};
    for (int k0 = 0; k0 < K; k0 += 32) {
        const f16* asrc = (k0 < 64) ? (P + (size_t)(m0 + mr) * 64 + k0)
                                    : (Q + (size_t)(m0 + mr) * 64 + (k0 - 64));
        v16h a = *(const v16h*)(asrc + 16 * h);
        #pragma unroll
        for (int t = 0; t < 4; ++t) {
            v16h bv = *(const v16h*)(WT + (size_t)(16 * t + mr) * K + k0 + 16 * h);
            acc[t] = __builtin_amdgcn_wmma_f32_16x16x32_f16(false, a, false, bv,
                                                            (short)0, acc[t], false, false);
        }
    }
    #pragma unroll
    for (int t = 0; t < 4; ++t) {
        #pragma unroll
        for (int r = 0; r < 8; ++r) {
            int row = m0 + r + 8 * h, col = 16 * t + mr;
            float v = acc[t][r];
            if (relu) v = fmaxf(v, 0.0f);
            if (out32) out32[(size_t)row * 64 + col] = v;
            if (out16) out16[(size_t)row * 64 + col] = (f16)v;
            if (out16T) {
                int b = row >> 10, n = row & (NN - 1);
                out16T[((size_t)b * NF + col) * NN + n] = (f16)v;
            }
        }
    }
}

// ------- agg16[b,i,:] = (Wfull @ x)[b,i,:] / norm[b,i]; block structure of Wfull -------
// A: rows i<512 -> w16T[b][i][k] ; rows >=512 -> w16[b][i-512][k]  (both k-contiguous)
// B: xT16[b][n][xoff + k]  (k-contiguous, col-major)
__global__ void k_agg(const f16* __restrict__ w16, const f16* __restrict__ w16T,
                      const f16* __restrict__ xT16, const float* __restrict__ norms,
                      f16* __restrict__ agg16)
{
    const int lane = threadIdx.x & 31;
    const int h = lane >> 4, mr = lane & 15;
    const int b = blockIdx.z;
    const int i0 = (blockIdx.x * 4 + (threadIdx.x >> 5)) * 16;
    const bool upper = (i0 < NU);
    const int xoff = upper ? NU : 0;
    const f16* asrc0 = upper ? (w16T + (size_t)b * NU * NU + (size_t)(i0 + mr) * NU)
                             : (w16 + (size_t)b * NU * NU + (size_t)(i0 - NU + mr) * NU);
    const f16* bbase = xT16 + (size_t)b * NF * NN + xoff;
    v8f acc[4] = {{}, {}, {}, {}};
    for (int k0 = 0; k0 < NU; k0 += 32) {
        v16h a = *(const v16h*)(asrc0 + k0 + 16 * h);
        #pragma unroll
        for (int t = 0; t < 4; ++t) {
            v16h bv = *(const v16h*)(bbase + (size_t)(16 * t + mr) * NN + k0 + 16 * h);
            acc[t] = __builtin_amdgcn_wmma_f32_16x16x32_f16(false, a, false, bv,
                                                            (short)0, acc[t], false, false);
        }
    }
    #pragma unroll
    for (int t = 0; t < 4; ++t) {
        #pragma unroll
        for (int r = 0; r < 8; ++r) {
            int row = i0 + r + 8 * h;
            float inv = 1.0f / norms[b * NN + row];
            agg16[((size_t)b * NN + row) * NF + 16 * t + mr] = (f16)(acc[t][r] * inv);
        }
    }
}

extern "C" void kernel_launch(void* const* d_in, const int* in_sizes, int n_in,
                              void* d_out, int out_size, void* d_ws, size_t ws_size,
                              hipStream_t stream) {
    (void)in_sizes; (void)n_in; (void)out_size; (void)ws_size;
    const float* nf    = (const float*)d_in[0];   // (2,1024,1)
    const float* adj   = (const float*)d_in[1];   // (2,1024,1024)
    const float* wts   = (const float*)d_in[2];   // (2,512,512)
    const float* Wi    = (const float*)d_in[3];   // (1,64)
    const float* bi    = (const float*)d_in[4];   // (64,)
    const float* Wee   = (const float*)d_in[5];   // (2,63)
    const float* Wef   = (const float*)d_in[6];   // (64,64)
    const float* Wmsg  = (const float*)d_in[7];   // (4,128,64)
    const float* Wupd  = (const float*)d_in[8];   // (4,128,64)
    float* out = (float*)d_out;                   // (2,1024,64)

    const int M = NB * NN;                        // 2048
    const int SZ = M * NF;                        // 131072

    size_t off = 0;
    char* base = (char*)d_ws;
    auto alloc = [&](size_t bytes) -> char* {
        char* p = base + off;
        off = (off + bytes + 255) & ~(size_t)255;
        return p;
    };
    float* norms   = (float*)alloc(M * 4);
    float* normmax = (float*)alloc(4);
    f16* x16A  = (f16*)alloc(SZ * 2);
    f16* x16B  = (f16*)alloc(SZ * 2);
    f16* xT16A = (f16*)alloc(SZ * 2);
    f16* xT16B = (f16*)alloc(SZ * 2);
    f16* eein  = (f16*)alloc(SZ * 2);
    f16* eemb  = (f16*)alloc(SZ * 2);
    f16* agg   = (f16*)alloc(SZ * 2);
    f16* msg   = (f16*)alloc(SZ * 2);
    f16* w16   = (f16*)alloc((size_t)NB * NU * NU * 2);
    f16* w16T  = (f16*)alloc((size_t)NB * NU * NU * 2);
    f16* WefT  = (f16*)alloc(4096 * 2);
    f16* WmsgT = (f16*)alloc(32768 * 2);
    f16* WupdT = (f16*)alloc(32768 * 2);

    k_prep_w<<<(NB * NU * NU + 255) / 256, 256, 0, stream>>>(wts, w16, w16T);
    k_prep_small<<<(4096 + 65536 + 255) / 256, 256, 0, stream>>>(Wef, Wmsg, Wupd,
                                                                 WefT, WmsgT, WupdT);
    k_x0<<<(SZ + 255) / 256, 256, 0, stream>>>(nf, Wi, bi, x16A, xT16A);
    k_norm<<<M / 256, 256, 0, stream>>>(adj, norms);
    k_normmax<<<1, 256, 0, stream>>>(norms, normmax);
    k_edgesum<<<M, 64, 0, stream>>>(adj, wts, nf, Wee, norms, normmax, eein);

    // edge_emb = relu(eein @ W_edge_feat)
    k_gemm_concat<<<dim3(M / 64), 128, 0, stream>>>(eein, nullptr, WefT,
                                                    nullptr, eemb, nullptr, 1);

    f16* xcur = x16A;  f16* xnext = x16B;
    f16* xTcur = xT16A; f16* xTnext = xT16B;
    for (int l = 0; l < NL; ++l) {
        k_agg<<<dim3(NN / 64, 1, NB), 128, 0, stream>>>(w16, w16T, xTcur, norms, agg);
        k_gemm_concat<<<dim3(M / 64), 128, 0, stream>>>(
            agg, eemb, WmsgT + (size_t)l * 8192, nullptr, msg, nullptr, 1);
        if (l == NL - 1) {
            k_gemm_concat<<<dim3(M / 64), 128, 0, stream>>>(
                xcur, msg, WupdT + (size_t)l * 8192, out, nullptr, nullptr, 0);
        } else {
            k_gemm_concat<<<dim3(M / 64), 128, 0, stream>>>(
                xcur, msg, WupdT + (size_t)l * 8192, nullptr, xnext, xTnext, 1);
        }
        f16* t;
        t = xcur; xcur = xnext; xnext = t;
        t = xTcur; xTcur = xTnext; xTnext = t;
    }
}
